// EdgeScoringNet_52097953300921
// MI455X (gfx1250) — matrix operand order repaired and checked by reading, not compiled, not run on hardware
//
#include <hip/hip_runtime.h>
#include <hip/hip_bf16.h>
#include <stdint.h>

#define N_EDGES 500000
#define FEAT 128
#define HID 256

typedef __attribute__((ext_vector_type(16))) __bf16 v16bf;
typedef __attribute__((ext_vector_type(8)))  float  v8f;

#define HAS_ASYNC_LDS __has_builtin(__builtin_amdgcn_global_load_async_to_lds_b128)

__device__ __forceinline__ void wait_async_zero() {
#if __has_builtin(__builtin_amdgcn_s_wait_asynccnt)
    __builtin_amdgcn_s_wait_asynccnt(0);
#else
    asm volatile("s_wait_asynccnt 0x0" ::: "memory");
#endif
}

__device__ __forceinline__ unsigned short f2bf(float f) {
    unsigned u = __builtin_bit_cast(unsigned, f);
    u += 0x7FFFu + ((u >> 16) & 1u);            // round-to-nearest-even
    return (unsigned short)(u >> 16);
}

// Repack W1 [HID=256, K=256] f32 into bf16 "B-fragment-native" order:
// flat index q = j + 16*(lane + 32*(kstep + 8*ntile))
// lane holds column n = ntile*16 + lane%16, K = kstep*32 + (j/8)*16 + (lane/16)*8 + j%8
__global__ void prep_w1_kernel(const float* __restrict__ W1,
                               unsigned short* __restrict__ Bpack) {
    int q    = blockIdx.x * 256 + threadIdx.x;   // 0..65535
    int j    = q & 15;
    int lane = (q >> 4) & 31;
    int kq   = (q >> 9) & 7;
    int nt   = q >> 12;                          // 0..15
    int n = nt * 16 + (lane & 15);
    int k = kq * 32 + (j >> 3) * 16 + (lane >> 4) * 8 + (j & 7);
    Bpack[q] = f2bf(W1[n * 256 + k]);
}

__global__ __launch_bounds__(256)
void edge_score_kernel(const float* __restrict__ mvc,
                       const int*   __restrict__ eidx,
                       const float* __restrict__ b1,
                       const float* __restrict__ W2,
                       const float* __restrict__ b2,
                       const uint4* __restrict__ Bg,
                       float*       __restrict__ out) {
    __shared__ uint4 sB[2][2048];                // 2 x 32 KB double buffer
    const int tid  = threadIdx.x;
    const int lane = tid & 31;
    const int wave = tid >> 5;
    const int half = lane >> 4;
    const int row  = lane & 15;

    const int e0 = (blockIdx.x * 8 + wave) * 16;
    int e  = e0 + row;
    int ec = e < N_EDGES ? e : N_EDGES - 1;      // clamp: keep EXEC uniform
    const int src = eidx[ec];
    const int dst = eidx[N_EDGES + ec];

    // stage one 32 KB chunk of packed W1 into LDS buffer `buf`
    auto stage = [&](int c, int buf) {
#if HAS_ASYNC_LDS
        typedef __attribute__((__vector_size__(4 * sizeof(int)))) int asy_v4i;
#pragma unroll
        for (int i = 0; i < 8; ++i) {
            const uint4* gp = &Bg[c * 2048 + i * 256 + tid];
            uint4*       lp = &sB[buf][i * 256 + tid];
            __builtin_amdgcn_global_load_async_to_lds_b128(
                (__attribute__((address_space(1))) asy_v4i*)gp,
                (__attribute__((address_space(3))) asy_v4i*)lp, 0, 0);
        }
#else
#pragma unroll
        for (int i = 0; i < 8; ++i)
            sB[buf][i * 256 + tid] = Bg[c * 2048 + i * 256 + tid];
        if (c < 3)
            __builtin_prefetch((const char*)Bg + (size_t)(c + 1) * 32768 +
                               (size_t)tid * 128, 0, 0);
#endif
    };
    auto finish_stage = [&]() {
#if HAS_ASYNC_LDS
        wait_async_zero();
#endif
    };

    // ---- A fragments: 16 edges x 256 (start|end feats), bf16, in registers.
    // ISA A-layout (16-bit, 16x32): lanes 0-15 slots = K 0..7 | 16..23,
    // lanes 16-31 slots = K 8..15 | 24..31  -> two contiguous 8-float chunks.
    v16bf a[8];
#pragma unroll
    for (int kk = 0; kk < 8; ++kk) {
        float x[16];
#pragma unroll
        for (int c = 0; c < 2; ++c) {
            int g    = kk * 32 + c * 16 + half * 8;   // global K index of chunk
            int node = (g < FEAT) ? src : dst;
            const float4* p = (const float4*)(mvc + (size_t)node * FEAT + (g & (FEAT - 1)));
            float4 u = p[0], v = p[1];
            x[c * 8 + 0] = u.x; x[c * 8 + 1] = u.y;
            x[c * 8 + 2] = u.z; x[c * 8 + 3] = u.w;
            x[c * 8 + 4] = v.x; x[c * 8 + 5] = v.y;
            x[c * 8 + 6] = v.z; x[c * 8 + 7] = v.w;
        }
        v16bf t;
#pragma unroll
        for (int i = 0; i < 16; ++i) t[i] = (__bf16)x[i];   // hw v_cvt_pk_bf16_f32
        a[kk] = t;
    }

    float p0[8], p1[8];
#pragma unroll
    for (int v = 0; v < 8; ++v) { p0[v] = 0.f; p1[v] = 0.f; }

    stage(0, 0);
    finish_stage();
    __syncthreads();

    for (int c = 0; c < 4; ++c) {                // 4 chunks of 4 N-tiles
        const int buf = c & 1;
        if (c < 3) stage(c + 1, buf ^ 1);        // overlap with WMMA below

#pragma unroll
        for (int nt = 0; nt < 4; ++nt) {
            const int ntile = c * 4 + nt;
            v8f acc = {};
#pragma unroll
            for (int kk = 0; kk < 8; ++kk) {
                int idx = ((nt * 8 + kk) * 32 + lane) * 2;
                uint4 lo = sB[buf][idx], hi = sB[buf][idx + 1];
                v16bf b;
                {
                    typedef __attribute__((ext_vector_type(8))) unsigned int v8u;
                    v8u t;
                    t[0] = lo.x; t[1] = lo.y; t[2] = lo.z; t[3] = lo.w;
                    t[4] = hi.x; t[5] = hi.y; t[6] = hi.z; t[7] = hi.w;
                    b = __builtin_bit_cast(v16bf, t);
                }
                acc = __builtin_amdgcn_wmma_f32_16x16x32_bf16(
                          false, a[kk], false, b, (short)0, acc, false, false);
            }
            // bias + relu + fc2 partials (C/D layout: VGPR v -> M=v+8*half, N=row)
            const int   ng   = ntile * 16 + row;
            const float bias = b1[ng];
            const float w20  = W2[ng];
            const float w21  = W2[HID + ng];
#pragma unroll
            for (int v = 0; v < 8; ++v) {
                float h = acc[v] + bias;
                h = h > 0.f ? h : 0.f;
                p0[v] = fmaf(h, w20, p0[v]);
                p1[v] = fmaf(h, w21, p1[v]);
            }
        }
        finish_stage();
        __syncthreads();
    }

    // reduce fc2 partials across the 16 columns of each half-group
#pragma unroll
    for (int v = 0; v < 8; ++v) {
#pragma unroll
        for (int m = 1; m < 16; m <<= 1) {
            p0[v] += __shfl_xor(p0[v], m, 16);
            p1[v] += __shfl_xor(p1[v], m, 16);
        }
    }

    if (row == 0) {                              // lanes 0 and 16 write results
        const float bb0 = b2[0], bb1 = b2[1];
#pragma unroll
        for (int v = 0; v < 8; ++v) {
            int eo = e0 + v + 8 * half;
            if (eo < N_EDGES) {
                out[2 * eo + 0] = p0[v] + bb0;
                out[2 * eo + 1] = p1[v] + bb1;
            }
        }
    }
}

extern "C" void kernel_launch(void* const* d_in, const int* in_sizes, int n_in,
                              void* d_out, int out_size, void* d_ws, size_t ws_size,
                              hipStream_t stream) {
    const float* mvc = (const float*)d_in[0];
    const int*   ei  = (const int*)d_in[1];
    const float* W1  = (const float*)d_in[2];
    const float* b1  = (const float*)d_in[3];
    const float* W2  = (const float*)d_in[4];
    const float* b2  = (const float*)d_in[5];
    float* out = (float*)d_out;
    unsigned short* Bpack = (unsigned short*)d_ws;   // 128 KB used

    prep_w1_kernel<<<256, 256, 0, stream>>>(W1, Bpack);

    int nblocks = (N_EDGES + 127) / 128;
    edge_score_kernel<<<nblocks, 256, 0, stream>>>(mvc, ei, b1, W2, b2,
                                                   (const uint4*)Bpack, out);
}
